// Model_3839700762834
// MI455X (gfx1250) — matrix operand order repaired and checked by reading, not compiled, not run on hardware
//
#include <hip/hip_runtime.h>
#include <stdint.h>

// Pixel-RNN (modReLU) on MI455X via f16 WMMA, f32 accumulation.
// z^T = W_hh (A, resident in VGPRs) x h^T (B) + (W_ih*x) (C), two 16x16x32
// WMMAs per 16-batch tile per step; D->B conversion is per-lane f16 packs +
// one half-wave cross-row permute (v_permlanex16, pure VALU — keeps the
// 784-step serial chain off the LDS pipeline) + cndmask selects.

typedef __attribute__((ext_vector_type(16))) _Float16 v16h;
typedef __attribute__((ext_vector_type(8)))  float    v8f;

#define HIDDEN 30
#define SEQ    784
#define NCLS   10
#define BATCH  16384
#define TILES  2                    // batch tiles per wave (ILP for WMMA chain)
#define WAVES  4
#define BLOCK  (WAVES * 32)
#define ROWS_PER_BLOCK (WAVES * TILES * 16)

union BMat { v16h h; uint32_t u[8]; };

__device__ __forceinline__ uint32_t pack_h2(float a, float b) {
  auto t = __builtin_amdgcn_cvt_pkrtz(a, b);      // v_cvt_pk_rtz_f16_f32
  return __builtin_bit_cast(uint32_t, t);
}

// lane L <-> lane L^16 swap. VALU cross-row permute (no LDS pipeline, no
// DScnt wait on the recurrence critical path).
__device__ __forceinline__ uint32_t xswap16(uint32_t v) {
#if __has_builtin(__builtin_amdgcn_permlanex16)
  return (uint32_t)__builtin_amdgcn_permlanex16(
      (int)v, (int)v, 0x76543210, 0xfedcba98, false, false);
#else
  return __shfl_xor(v, 16, 32);
#endif
}

__device__ __forceinline__ float modrelu(float z, float b) {
  float a = __builtin_fabsf(z) + b;               // v_add_f32 with |src|
  a = __builtin_fmaxf(a, 0.0f);
  return __builtin_copysignf(a, z);               // sign(z)*relu(|z|+b)
}

__global__ __launch_bounds__(BLOCK) void irnn_wmma_kernel(
    const float* __restrict__ inputs,   // [BATCH, SEQ]
    const float* __restrict__ W_ih,     // [HIDDEN]
    const float* __restrict__ W_hh,     // [HIDDEN, HIDDEN]
    const float* __restrict__ b_mod,    // [HIDDEN]
    const float* __restrict__ W_lin,    // [NCLS, HIDDEN]
    const float* __restrict__ b_lin,    // [NCLS]
    float* __restrict__ out)            // [BATCH, NCLS]
{
  const int lane = threadIdx.x & 31;
  const int wave = threadIdx.x >> 5;
  const int g    = lane >> 4;          // half-wave group
  const int col  = lane & 15;          // batch column within tile / A row

  // ---- A tiles (constant for all 784 steps). k order used here (contiguous, same as B):
  // lane group 0: element e -> K = e; group 1: K = e+16.
  v16h a1, a2, al;
  #pragma unroll
  for (int e = 0; e < 16; ++e) {
    const int k  = e + 16 * g;                       // K index, 0..31
    const int m2 = 16 + col;
    a1[e] = (k < HIDDEN) ? (_Float16)W_hh[col * HIDDEN + k] : (_Float16)0.0f;
    a2[e] = (m2 < HIDDEN && k < HIDDEN) ? (_Float16)W_hh[m2 * HIDDEN + k]
                                        : (_Float16)0.0f;
    al[e] = (col < NCLS && k < HIDDEN) ? (_Float16)W_lin[col * HIDDEN + k]
                                       : (_Float16)0.0f;
  }

  // Per-lane broadcast scalars in C/D layout (row M = r + 8g).
  float wih1[8], wih2[8], bm1[8], bm2[8], bl[8];
  #pragma unroll
  for (int r = 0; r < 8; ++r) {
    const int m1 = r + 8 * g;          // 0..15  (tile 1 rows)
    const int m2 = 16 + m1;            // 16..31 (tile 2 rows)
    wih1[r] = W_ih[m1];
    bm1[r]  = b_mod[m1];
    wih2[r] = (m2 < HIDDEN) ? W_ih[m2]  : 0.0f;
    bm2[r]  = (m2 < HIDDEN) ? b_mod[m2] : 0.0f;
    bl[r]   = (m1 < NCLS)   ? b_lin[m1] : 0.0f;
  }

  // ---- per-tile state: B matrix (h^T, f16), input row pointers
  BMat bmat[TILES];
  const float4* xp[TILES];
  int base[TILES];
  #pragma unroll
  for (int i = 0; i < TILES; ++i) {
    #pragma unroll
    for (int j = 0; j < 8; ++j) bmat[i].u[j] = 0u;   // h0 = 0
    base[i] = blockIdx.x * ROWS_PER_BLOCK + (wave * TILES + i) * 16;
    xp[i] = reinterpret_cast<const float4*>(inputs + (size_t)(base[i] + col) * SEQ);
  }

  float4 cur[TILES];
  #pragma unroll
  for (int i = 0; i < TILES; ++i) cur[i] = xp[i][0];

  for (int it = 0; it < SEQ / 4; ++it) {
    float4 nxt[TILES];
    if (it + 1 < SEQ / 4) {                           // software-pipelined x load
      #pragma unroll
      for (int i = 0; i < TILES; ++i) nxt[i] = xp[i][it + 1];
    }
    #pragma unroll
    for (int s = 0; s < 4; ++s) {
      #pragma unroll
      for (int i = 0; i < TILES; ++i) {
        const float x = (s == 0) ? cur[i].x : (s == 1) ? cur[i].y
                      : (s == 2) ? cur[i].z : cur[i].w;
        // C = x * W_ih (input projection folded into the accumulator input)
        v8f c1, c2;
        #pragma unroll
        for (int r = 0; r < 8; ++r) { c1[r] = wih1[r] * x; c2[r] = wih2[r] * x; }

        v8f d1 = __builtin_amdgcn_wmma_f32_16x16x32_f16(
            false, a1, false, bmat[i].h, (short)0, c1, false, false);
        v8f d2 = __builtin_amdgcn_wmma_f32_16x16x32_f16(
            false, a2, false, bmat[i].h, (short)0, c2, false, false);

        #pragma unroll
        for (int r = 0; r < 8; ++r) {
          d1[r] = modrelu(d1[r], bm1[r]);
          d2[r] = modrelu(d2[r], bm2[r]);
        }

        // D (z^T) -> B (h^T): pack f32 pairs to f16, cross-row swap, select.
        uint32_t p1[4], p2[4];
        #pragma unroll
        for (int j = 0; j < 4; ++j) {
          p1[j] = pack_h2(d1[2 * j], d1[2 * j + 1]);  // hidden (0..7)+8g
          p2[j] = pack_h2(d2[2 * j], d2[2 * j + 1]);  // hidden (16..23)+8g
        }
        #pragma unroll
        for (int j = 0; j < 4; ++j) {
          const uint32_t q1 = xswap16(p1[j]);
          const uint32_t q2 = xswap16(p2[j]);
          // B lanes 0-15 hold K=0..15 (p1|q1); lanes 16-31 hold K=16..31 (q2|p2)
          bmat[i].u[j]     = (g == 0) ? p1[j] : q2;
          bmat[i].u[4 + j] = (g == 0) ? q1    : p2[j];
        }
      }
    }
    if (it + 1 < SEQ / 4) {
      #pragma unroll
      for (int i = 0; i < TILES; ++i) cur[i] = nxt[i];
    }
  }

  // ---- classifier head: out^T = W_lin x h^T + b_lin, one WMMA per tile
  #pragma unroll
  for (int i = 0; i < TILES; ++i) {
    v8f cl;
    #pragma unroll
    for (int r = 0; r < 8; ++r) cl[r] = bl[r];
    v8f d = __builtin_amdgcn_wmma_f32_16x16x32_f16(
        false, al, false, bmat[i].h, (short)0, cl, false, false);
    const int n = base[i] + col;
    float* o = out + (size_t)n * NCLS;
    if (g == 0) {                       // classes 0..7 in lanes 0-15
      #pragma unroll
      for (int r = 0; r < 8; ++r) o[r] = d[r];
    } else {                            // classes 8,9 in lanes 16-31 (r=0,1)
      o[8] = d[0];
      o[9] = d[1];
    }
  }
}

extern "C" void kernel_launch(void* const* d_in, const int* in_sizes, int n_in,
                              void* d_out, int out_size, void* d_ws, size_t ws_size,
                              hipStream_t stream) {
  const float* inputs = (const float*)d_in[0];
  const float* W_ih   = (const float*)d_in[1];
  const float* W_hh   = (const float*)d_in[2];
  const float* b_mod  = (const float*)d_in[3];
  const float* W_lin  = (const float*)d_in[4];
  const float* b_lin  = (const float*)d_in[5];
  float* out = (float*)d_out;

  dim3 grid(BATCH / ROWS_PER_BLOCK);   // 128 blocks
  dim3 block(BLOCK);                   // 128 threads = 4 wave32
  hipLaunchKernelGGL(irnn_wmma_kernel, grid, block, 0, stream,
                     inputs, W_ih, W_hh, b_mod, W_lin, b_lin, out);
}
